// QuantDiffBTB_60601988547070
// MI455X (gfx1250) — matrix operand order repaired and checked by reading, not compiled, not run on hardware
//
#include <hip/hip_runtime.h>
#include <hip/hip_bf16.h>
#include <stdint.h>

// ---------------------------------------------------------------------------
// Types
// ---------------------------------------------------------------------------
typedef __attribute__((ext_vector_type(16))) __bf16 v16bf;
typedef __attribute__((ext_vector_type(8)))  float  v8f;

#define BM 64
#define BN 128
#define BK 32

// Problem constants (from the reference)
static constexpr int CB  = 8;      // batch
static constexpr int CS  = 1024;   // seq
static constexpr int CD  = 1152;   // model dim
static constexpr int CH  = 16;     // heads
static constexpr int CHD = 72;     // head dim
static constexpr int CHDP= 96;     // head dim padded to K-multiple of 32
static constexpr int CSE = 77;     // encoder seq
static constexpr int CSEP= 96;     // padded key count for cross PV reduction
static constexpr int CFF = 4608;
static constexpr int CBS = CB * CS;
static constexpr int CMC = 1024;   // FF row chunk

// ---------------------------------------------------------------------------
// CDNA5 async global->LDS copy (async DMA path, tracked by ASYNCcnt)
// ---------------------------------------------------------------------------
__device__ __forceinline__ void async_b128(const __bf16* g, __bf16* l) {
  asm volatile("global_load_async_to_lds_b128 %0, %1, off"
               :: "v"((unsigned)(uintptr_t)l),
                  "v"((unsigned long long)(uintptr_t)g)
               : "memory");
}
__device__ __forceinline__ void wait_async0() {
  asm volatile("s_wait_asynccnt 0" ::: "memory");
}

// ---------------------------------------------------------------------------
// Ordered-float encoding for atomic min/max on unsigned
// ---------------------------------------------------------------------------
__device__ __forceinline__ unsigned f2ord(float f) {
  unsigned u = __float_as_uint(f);
  return (u >> 31) ? ~u : (u | 0x80000000u);
}
__device__ __forceinline__ float ord2f(unsigned e) {
  unsigned u = (e >> 31) ? (e & 0x7FFFFFFFu) : ~e;
  return __uint_as_float(u);
}

// ---------------------------------------------------------------------------
// GEMM: C[M,N] = A[M,K] (bf16, row major) * Bt[N,K]^T (bf16, row major)
// 8 waves per block (wave32), block tile 64x128, each wave 16x64 (4 WMMA accs)
// Double-buffered LDS, async global->LDS staging, clamped (guard-free) loads.
// All fragments are loaded before the WMMA group so DS waits can be partial.
// EPI: 0 = f32 store, 1 = f32 resid+bias, 2 = bf16 store,
//      3 = PV: bf16 scatter into [B,S,H*HD] with col < N(=72)
// ---------------------------------------------------------------------------
struct GemmP {
  const __bf16* A;  long long aBatch; int lda;
  const __bf16* Bt; long long bBatch; int ldb;
  int M, N, K;
  float* Cf; const float* resid; const float* bias;
  long long cBatch; int ldc;
  __bf16* Cb;
  float alpha;
  int Hheads, pvB, Srows;
};

template <int EPI>
__global__ __launch_bounds__(256) void gemm_wmma(GemmP p) {
  __shared__ __attribute__((aligned(32))) __bf16 As[2][BM][BK];
  __shared__ __attribute__((aligned(32))) __bf16 Bs[2][BN][BK];

  const int bh = blockIdx.z;
  const __bf16* A  = p.A  + (long long)bh * p.aBatch;
  const __bf16* Bt = p.Bt + (long long)bh * p.bBatch;

  const int m0 = blockIdx.y * BM;
  const int n0 = blockIdx.x * BN;
  const int t    = threadIdx.x;
  const int lane = t & 31;
  const int wave = t >> 5;
  const int wm = wave & 3;        // 4 wave-rows
  const int wn = wave >> 2;       // 2 wave-cols
  const int half = lane >> 4;     // lane group
  const int l16  = lane & 15;

  v8f acc[4] = {};

  // Staging coordinates: A tile 64x32 (1 x b128/thread), B tile 128x32 (2).
  const int arow = t >> 2;             // 0..63
  const int acol = (t & 3) * 8;        // 0,8,16,24 halves
  // Clamp OOB rows to a valid row: garbage only feeds discarded C rows/cols.
  const int gmA = min(m0 + arow, p.M - 1);
  const int gn0 = min(n0 + arow, p.N - 1);
  const int gn1 = min(n0 + arow + 64, p.N - 1);
  const __bf16* aRow  = A  + (long long)gmA * p.lda + acol;
  const __bf16* bRow0 = Bt + (long long)gn0 * p.ldb + acol;
  const __bf16* bRow1 = Bt + (long long)gn1 * p.ldb + acol;

  // Preload K-step 0 into buffer 0 via async DMA.
  async_b128(aRow,  &As[0][arow][acol]);
  async_b128(bRow0, &Bs[0][arow][acol]);
  async_b128(bRow1, &Bs[0][arow + 64][acol]);
  wait_async0();
  __syncthreads();

  const int nk = p.K / BK;
  int cur = 0;
  for (int kt = 0; kt < nk; ++kt) {
    const int nxt = cur ^ 1;
    const bool more = (kt + 1) < nk;
    if (more) {
      const int k1 = (kt + 1) * BK;
      async_b128(aRow  + k1, &As[nxt][arow][acol]);
      async_b128(bRow0 + k1, &Bs[nxt][arow][acol]);
      async_b128(bRow1 + k1, &Bs[nxt][arow + 64][acol]);
    }

    // Load ALL fragments first so ds waits can be partial / overlapped.
    // A fragment: lane<16 -> M=l16, K 0..15; lane>=16 -> K 16..31 (contig)
    union { struct { uint4 lo, hi; } u; v16bf v; } af;
    af.u.lo = *reinterpret_cast<const uint4*>(&As[cur][wm * 16 + l16][half * 16]);
    af.u.hi = *reinterpret_cast<const uint4*>(&As[cur][wm * 16 + l16][half * 16 + 8]);
    // B fragments: lane<16 -> col=l16, K 0..15; lane>=16 -> K 16..31 (contig)
    const v16bf bf0 = *reinterpret_cast<const v16bf*>(
        &Bs[cur][wn * 64 + 0 * 16 + l16][half * 16]);
    const v16bf bf1 = *reinterpret_cast<const v16bf*>(
        &Bs[cur][wn * 64 + 1 * 16 + l16][half * 16]);
    const v16bf bf2 = *reinterpret_cast<const v16bf*>(
        &Bs[cur][wn * 64 + 2 * 16 + l16][half * 16]);
    const v16bf bf3 = *reinterpret_cast<const v16bf*>(
        &Bs[cur][wn * 64 + 3 * 16 + l16][half * 16]);

    acc[0] = __builtin_amdgcn_wmma_f32_16x16x32_bf16(
        false, af.v, false, bf0, (short)0, acc[0], false, false);
    acc[1] = __builtin_amdgcn_wmma_f32_16x16x32_bf16(
        false, af.v, false, bf1, (short)0, acc[1], false, false);
    acc[2] = __builtin_amdgcn_wmma_f32_16x16x32_bf16(
        false, af.v, false, bf2, (short)0, acc[2], false, false);
    acc[3] = __builtin_amdgcn_wmma_f32_16x16x32_bf16(
        false, af.v, false, bf3, (short)0, acc[3], false, false);

    if (more) wait_async0();
    __syncthreads();
    cur = nxt;
  }

  // ---- epilogue: VGPR i holds row (half*8 + i), lane low 4 bits hold col ----
#pragma unroll
  for (int j = 0; j < 4; j++) {
#pragma unroll
    for (int i = 0; i < 8; i++) {
      const int row = m0 + wm * 16 + half * 8 + i;
      const int col = n0 + wn * 64 + j * 16 + l16;
      const float val = acc[j][i] * p.alpha;
      if (EPI == 0) {
        if (row < p.M && col < p.N)
          p.Cf[(long long)bh * p.cBatch + (long long)row * p.ldc + col] = val;
      } else if (EPI == 1) {
        if (row < p.M && col < p.N) {
          const long long idx =
              (long long)bh * p.cBatch + (long long)row * p.ldc + col;
          const float r = p.resid ? p.resid[idx] : 0.0f;
          const float b = p.bias ? p.bias[col] : 0.0f;
          p.Cf[idx] = r + val + b;
        }
      } else if (EPI == 2) {
        if (row < p.M && col < p.N)
          p.Cb[(long long)bh * p.cBatch + (long long)row * p.ldc + col] =
              (__bf16)val;
      } else { // PV scatter into [B,S,H*HD]
        if (row < p.M && col < p.N) {
          const int h_ = bh % p.Hheads;
          const int b_ = p.pvB + bh / p.Hheads;
          const long long idx =
              ((long long)b_ * p.Srows + row) * (long long)p.ldc + h_ * p.N + col;
          p.Cb[idx] = (__bf16)val;
        }
      }
    }
  }
}

// ---------------------------------------------------------------------------
// Weight transpose + fp32 -> bf16:  W[K,N] -> Wt[N,K]
// ---------------------------------------------------------------------------
__global__ __launch_bounds__(256) void wtrans(const float* W, __bf16* Wt,
                                              int K, int N) {
  __shared__ float tile[32][33];
  const int k0 = blockIdx.y * 32, n0 = blockIdx.x * 32;
  const int tx = threadIdx.x & 31, ty = threadIdx.x >> 5;
  for (int i = ty; i < 32; i += 8) {
    const int k = k0 + i, n = n0 + tx;
    tile[i][tx] = (k < K && n < N) ? W[(long long)k * N + n] : 0.0f;
  }
  __syncthreads();
  for (int i = ty; i < 32; i += 8) {
    const int n = n0 + i, k = k0 + tx;
    if (n < N && k < K) Wt[(long long)n * K + k] = (__bf16)tile[tx][i];
  }
}

__global__ void f2bf(const float* x, __bf16* y, long long n) {
  for (long long i = blockIdx.x * 256ll + threadIdx.x; i < n;
       i += (long long)gridDim.x * 256ll)
    y[i] = (__bf16)x[i];
}

// ---------------------------------------------------------------------------
// LayerNorm (fp32 in -> bf16 out), one block per row of D
// ---------------------------------------------------------------------------
__global__ __launch_bounds__(256) void ln_bf16(const float* x, const float* g,
                                               const float* b, __bf16* y, int D) {
  __shared__ float red[256];
  const long long row = blockIdx.x;
  const float* xr = x + row * (long long)D;
  float s = 0.0f, s2 = 0.0f;
  for (int i = threadIdx.x; i < D; i += 256) {
    const float v = xr[i];
    s += v; s2 += v * v;
  }
  red[threadIdx.x] = s; __syncthreads();
  for (int o = 128; o > 0; o >>= 1) {
    if (threadIdx.x < o) red[threadIdx.x] += red[threadIdx.x + o];
    __syncthreads();
  }
  const float mean = red[0] / D;
  __syncthreads();
  red[threadIdx.x] = s2; __syncthreads();
  for (int o = 128; o > 0; o >>= 1) {
    if (threadIdx.x < o) red[threadIdx.x] += red[threadIdx.x + o];
    __syncthreads();
  }
  const float var = red[0] / D - mean * mean;
  const float inv = rsqrtf(var + 1e-5f);
  __bf16* yr = y + row * (long long)D;
  for (int i = threadIdx.x; i < D; i += 256)
    yr[i] = (__bf16)((xr[i] - mean) * inv * g[i] + b[i]);
}

// ---------------------------------------------------------------------------
// Quantizer statistic kernels
// ---------------------------------------------------------------------------
__global__ void stat_init(unsigned* s) {
  const int t = threadIdx.x;
  if (t < 8) s[t] = (t == 0 || t == 2 || t == 4) ? 0xFFFFFFFFu : 0u;
}

__global__ __launch_bounds__(256) void minmax_k(const float* x, long long n,
                                                unsigned* out) {
  float lo = INFINITY, hi = -INFINITY;
  for (long long i = blockIdx.x * 256ll + threadIdx.x; i < n;
       i += (long long)gridDim.x * 256ll) {
    const float v = x[i];
    lo = fminf(lo, v); hi = fmaxf(hi, v);
  }
  __shared__ float smn[256], smx[256];
  smn[threadIdx.x] = lo; smx[threadIdx.x] = hi; __syncthreads();
  for (int o = 128; o > 0; o >>= 1) {
    if (threadIdx.x < o) {
      smn[threadIdx.x] = fminf(smn[threadIdx.x], smn[threadIdx.x + o]);
      smx[threadIdx.x] = fmaxf(smx[threadIdx.x], smx[threadIdx.x + o]);
    }
    __syncthreads();
  }
  if (threadIdx.x == 0) {
    atomicMin(&out[0], f2ord(smn[0]));
    atomicMax(&out[1], f2ord(smx[0]));
  }
}

__device__ __forceinline__ float fakequant(float v, float mn, float mx) {
  const float delta = fmaxf((mx - mn) * (1.0f / 255.0f), 1e-12f);
  const float zp = rintf(-mn / delta);
  const float q = fminf(fmaxf(rintf(v / delta) + zp, 0.0f), 255.0f);
  return (q - zp) * delta;
}

// x[B,S,D] fp32 -> out[B,H,S,HDP] bf16 quantized, head-dim padded with zeros
__global__ __launch_bounds__(256) void quant_pad_qk(const float* x,
                                                    const unsigned* st, int stB,
                                                    __bf16* out, int Bn, int S,
                                                    int Hh, int HD, int HDP,
                                                    int D) {
  const long long total = (long long)Bn * Hh * S * HDP;
  const float mn = ord2f(st[stB]), mx = ord2f(st[stB + 1]);
  for (long long idx = blockIdx.x * 256ll + threadIdx.x; idx < total;
       idx += (long long)gridDim.x * 256ll) {
    const int d = (int)(idx % HDP);
    long long r = idx / HDP;
    const int s = (int)(r % S); r /= S;
    const int h = (int)(r % Hh);
    const int b = (int)(r / Hh);
    float o = 0.0f;
    if (d < HD)
      o = fakequant(x[((long long)b * S + s) * D + h * HD + d], mn, mx);
    out[idx] = (__bf16)o;
  }
}

// v[B,Sk,D] fp32 -> vt[B,H,HD,SKP] bf16 quantized + transposed, keys padded
__global__ __launch_bounds__(256) void quant_v_t(const float* x,
                                                 const unsigned* st, int stB,
                                                 __bf16* out, int Bn, int Sk,
                                                 int SKP, int Hh, int HD, int D) {
  const long long total = (long long)Bn * Hh * HD * SKP;
  const float mn = ord2f(st[stB]), mx = ord2f(st[stB + 1]);
  for (long long idx = blockIdx.x * 256ll + threadIdx.x; idx < total;
       idx += (long long)gridDim.x * 256ll) {
    const int kk = (int)(idx % SKP);
    long long r = idx / SKP;
    const int d = (int)(r % HD); r /= HD;
    const int h = (int)(r % Hh);
    const int b = (int)(r / Hh);
    float o = 0.0f;
    if (kk < Sk)
      o = fakequant(x[((long long)b * Sk + kk) * D + h * HD + d], mn, mx);
    out[idx] = (__bf16)o;
  }
}

// ---------------------------------------------------------------------------
// Softmax row stats (max, sumexp) + global max-prob (= max over rows of 1/l)
// ---------------------------------------------------------------------------
__global__ __launch_bounds__(256) void sm_stats(const float* sc, int Sk, int ldS,
                                                float scale, float2* stats,
                                                long long stBase,
                                                unsigned* pmax) {
  const long long row = blockIdx.x;
  const float* sr = sc + row * (long long)ldS;
  __shared__ float red[256];
  float m = -INFINITY;
  for (int i = threadIdx.x; i < Sk; i += 256) m = fmaxf(m, sr[i] * scale);
  red[threadIdx.x] = m; __syncthreads();
  for (int o = 128; o > 0; o >>= 1) {
    if (threadIdx.x < o)
      red[threadIdx.x] = fmaxf(red[threadIdx.x], red[threadIdx.x + o]);
    __syncthreads();
  }
  m = red[0];
  __syncthreads();
  float l = 0.0f;
  for (int i = threadIdx.x; i < Sk; i += 256) l += expf(sr[i] * scale - m);
  red[threadIdx.x] = l; __syncthreads();
  for (int o = 128; o > 0; o >>= 1) {
    if (threadIdx.x < o) red[threadIdx.x] += red[threadIdx.x + o];
    __syncthreads();
  }
  if (threadIdx.x == 0) {
    l = red[0];
    stats[stBase + row] = make_float2(m, l);
    atomicMax(pmax, f2ord(1.0f / l));   // largest prob of this row
  }
}

// probs = exp(s*scale - m)/l, 8-bit always-zero quant, -> bf16 [rows, SKP]
__global__ __launch_bounds__(256) void quant_probs(const float* sc,
                                                   const float2* stats,
                                                   long long stBase,
                                                   const unsigned* pmax,
                                                   __bf16* pq, int Sk, int SKP,
                                                   int ldS, float scale) {
  const long long row = blockIdx.x;
  const float2 st = stats[stBase + row];
  const float delta = fmaxf(ord2f(*pmax) * (1.0f / 255.0f), 1e-20f);
  const float* sr = sc + row * (long long)ldS;
  __bf16* pr = pq + row * (long long)SKP;
  for (int i = threadIdx.x; i < SKP; i += 256) {
    float o = 0.0f;
    if (i < Sk) {
      const float p = expf(sr[i] * scale - st.x) / st.y;
      o = fminf(fmaxf(rintf(p / delta), 0.0f), 255.0f) * delta;
    }
    pr[i] = (__bf16)o;
  }
}

// ---------------------------------------------------------------------------
// GEGLU: proj[rows, 2*FF] fp32 (bias already added) -> g bf16 [rows, FF]
// ---------------------------------------------------------------------------
__global__ __launch_bounds__(256) void geglu_k(const float* proj, __bf16* g,
                                               int rows, int FFd) {
  const long long total = (long long)rows * FFd;
  for (long long idx = blockIdx.x * 256ll + threadIdx.x; idx < total;
       idx += (long long)gridDim.x * 256ll) {
    const long long r = idx / FFd;
    const int c = (int)(idx % FFd);
    const float a = proj[r * 2 * FFd + c];
    const float x = proj[r * 2 * FFd + FFd + c];
    const float gl = 0.5f * x * (1.0f + erff(x * 0.70710678118f));
    g[idx] = (__bf16)(a * gl);
  }
}

// ---------------------------------------------------------------------------
// Host orchestration
// ---------------------------------------------------------------------------
extern "C" void kernel_launch(void* const* d_in, const int* in_sizes, int n_in,
                              void* d_out, int out_size, void* d_ws,
                              size_t ws_size, hipStream_t stream) {
  (void)in_sizes; (void)n_in; (void)out_size; (void)ws_size;
  const float* hs   = (const float*)d_in[0];
  const float* enc  = (const float*)d_in[1];
  const float* ln1g = (const float*)d_in[2];
  const float* ln1b = (const float*)d_in[3];
  const float* Wq1  = (const float*)d_in[4];
  const float* Wk1  = (const float*)d_in[5];
  const float* Wv1  = (const float*)d_in[6];
  const float* Wo1  = (const float*)d_in[7];
  const float* bo1  = (const float*)d_in[8];
  const float* ln2g = (const float*)d_in[9];
  const float* ln2b = (const float*)d_in[10];
  const float* Wq2  = (const float*)d_in[11];
  const float* Wk2  = (const float*)d_in[12];
  const float* Wv2  = (const float*)d_in[13];
  const float* Wo2  = (const float*)d_in[14];
  const float* bo2  = (const float*)d_in[15];
  const float* ln3g = (const float*)d_in[16];
  const float* ln3b = (const float*)d_in[17];
  const float* Wff1 = (const float*)d_in[18];
  const float* bff1 = (const float*)d_in[19];
  const float* Wff2 = (const float*)d_in[20];
  const float* bff2 = (const float*)d_in[21];
  float* out = (float*)d_out;

  // ---- workspace bump allocator ----
  char* base = (char*)d_ws;
  size_t off = 0;
  auto alloc = [&](size_t bytes) -> void* {
    void* p = base + off;
    off = (off + bytes + 255) & ~(size_t)255;
    return p;
  };

  __bf16* Wq1t = (__bf16*)alloc((size_t)CD * CD * 2);
  __bf16* Wk1t = (__bf16*)alloc((size_t)CD * CD * 2);
  __bf16* Wv1t = (__bf16*)alloc((size_t)CD * CD * 2);
  __bf16* Wo1t = (__bf16*)alloc((size_t)CD * CD * 2);
  __bf16* Wq2t = (__bf16*)alloc((size_t)CD * CD * 2);
  __bf16* Wk2t = (__bf16*)alloc((size_t)CD * CD * 2);
  __bf16* Wv2t = (__bf16*)alloc((size_t)CD * CD * 2);
  __bf16* Wo2t = (__bf16*)alloc((size_t)CD * CD * 2);
  __bf16* W1t  = (__bf16*)alloc((size_t)(2 * CFF) * CD * 2);
  __bf16* W2t  = (__bf16*)alloc((size_t)CD * CFF * 2);
  __bf16* lnbuf = (__bf16*)alloc((size_t)CBS * CD * 2);
  __bf16* encb  = (__bf16*)alloc((size_t)CB * CSE * CD * 2);
  float* qf = (float*)alloc((size_t)CBS * CD * 4);
  float* kf = (float*)alloc((size_t)CBS * CD * 4);
  float* vf = (float*)alloc((size_t)CBS * CD * 4);
  __bf16* qp = (__bf16*)alloc((size_t)CB * CH * CS * CHDP * 2);
  __bf16* kp = (__bf16*)alloc((size_t)CB * CH * CS * CHDP * 2);
  __bf16* vt = (__bf16*)alloc((size_t)CB * CH * CHD * CS * 2);
  float* scores = (float*)alloc((size_t)CH * CS * CS * 4);       // per-b self
  __bf16* pq = (__bf16*)alloc((size_t)CH * CS * CS * 2);         // per-b self
  float2* stats = (float2*)alloc((size_t)CB * CH * CS * sizeof(float2));
  __bf16* attn = (__bf16*)alloc((size_t)CBS * CD * 2);
  float* proj = (float*)alloc((size_t)CMC * 2 * CFF * 4);
  __bf16* gbuf = (__bf16*)alloc((size_t)CMC * CFF * 2);
  unsigned* st = (unsigned*)alloc(64 * 4);

  auto gemm = [&](int epi, const __bf16* A, long long aB, int lda,
                  const __bf16* Bt, long long bB, int ldb, int M, int N, int K,
                  int batch, float* Cf, const float* resid, const float* bias,
                  long long cB, int ldc, __bf16* Cb, int pvB, int Srows) {
    GemmP p{};
    p.A = A; p.aBatch = aB; p.lda = lda;
    p.Bt = Bt; p.bBatch = bB; p.ldb = ldb;
    p.M = M; p.N = N; p.K = K;
    p.Cf = Cf; p.resid = resid; p.bias = bias;
    p.cBatch = cB; p.ldc = ldc; p.Cb = Cb;
    p.alpha = 1.0f; p.Hheads = CH; p.pvB = pvB; p.Srows = Srows;
    dim3 g((N + BN - 1) / BN, (M + BM - 1) / BM, batch);
    switch (epi) {
      case 0: gemm_wmma<0><<<g, 256, 0, stream>>>(p); break;
      case 1: gemm_wmma<1><<<g, 256, 0, stream>>>(p); break;
      case 2: gemm_wmma<2><<<g, 256, 0, stream>>>(p); break;
      default: gemm_wmma<3><<<g, 256, 0, stream>>>(p); break;
    }
  };
  auto wt = [&](const float* W, int K, int N, __bf16* dst) {
    dim3 g((N + 31) / 32, (K + 31) / 32);
    wtrans<<<g, 256, 0, stream>>>(W, dst, K, N);
  };

  const float scale = 0.117851130f;  // 1/sqrt(72)

  // residual stream lives in d_out
  hipMemcpyAsync(out, hs, (size_t)CBS * CD * 4, hipMemcpyDeviceToDevice, stream);

  // weights -> bf16 transposed [N,K]
  wt(Wq1, CD, CD, Wq1t); wt(Wk1, CD, CD, Wk1t);
  wt(Wv1, CD, CD, Wv1t); wt(Wo1, CD, CD, Wo1t);
  wt(Wq2, CD, CD, Wq2t); wt(Wk2, CD, CD, Wk2t);
  wt(Wv2, CD, CD, Wv2t); wt(Wo2, CD, CD, Wo2t);
  wt(Wff1, CD, 2 * CFF, W1t); wt(Wff2, CFF, CD, W2t);
  f2bf<<<4096, 256, 0, stream>>>(enc, encb, (long long)CB * CSE * CD);

  // ========================= self-attention =========================
  ln_bf16<<<CBS, 256, 0, stream>>>(out, ln1g, ln1b, lnbuf, CD);
  gemm(0, lnbuf, 0, CD, Wq1t, 0, CD, CBS, CD, CD, 1, qf, nullptr, nullptr, 0, CD, nullptr, 0, 0);
  gemm(0, lnbuf, 0, CD, Wk1t, 0, CD, CBS, CD, CD, 1, kf, nullptr, nullptr, 0, CD, nullptr, 0, 0);
  gemm(0, lnbuf, 0, CD, Wv1t, 0, CD, CBS, CD, CD, 1, vf, nullptr, nullptr, 0, CD, nullptr, 0, 0);
  stat_init<<<1, 32, 0, stream>>>(st);
  minmax_k<<<2048, 256, 0, stream>>>(qf, (long long)CBS * CD, st + 0);
  minmax_k<<<2048, 256, 0, stream>>>(kf, (long long)CBS * CD, st + 2);
  minmax_k<<<2048, 256, 0, stream>>>(vf, (long long)CBS * CD, st + 4);
  {
    long long tq = (long long)CB * CH * CS * CHDP;
    quant_pad_qk<<<(unsigned)((tq + 255) / 256), 256, 0, stream>>>(
        qf, st, 0, qp, CB, CS, CH, CHD, CHDP, CD);
    quant_pad_qk<<<(unsigned)((tq + 255) / 256), 256, 0, stream>>>(
        kf, st, 2, kp, CB, CS, CH, CHD, CHDP, CD);
    long long tv = (long long)CB * CH * CHD * CS;
    quant_v_t<<<(unsigned)((tv + 255) / 256), 256, 0, stream>>>(
        vf, st, 4, vt, CB, CS, CS, CH, CHD, CD);
  }
  // pass 1: per-batch scores -> row stats + global prob max
  for (int b = 0; b < CB; b++) {
    const __bf16* qpb = qp + (size_t)b * CH * CS * CHDP;
    const __bf16* kpb = kp + (size_t)b * CH * CS * CHDP;
    gemm(0, qpb, (long long)CS * CHDP, CHDP, kpb, (long long)CS * CHDP, CHDP,
         CS, CS, CHDP, CH, scores, nullptr, nullptr, (long long)CS * CS, CS,
         nullptr, 0, 0);
    sm_stats<<<CH * CS, 256, 0, stream>>>(scores, CS, CS, scale, stats,
                                          (long long)b * CH * CS, st + 6);
  }
  // pass 2: recompute scores, quantize probs, PV
  for (int b = 0; b < CB; b++) {
    const __bf16* qpb = qp + (size_t)b * CH * CS * CHDP;
    const __bf16* kpb = kp + (size_t)b * CH * CS * CHDP;
    gemm(0, qpb, (long long)CS * CHDP, CHDP, kpb, (long long)CS * CHDP, CHDP,
         CS, CS, CHDP, CH, scores, nullptr, nullptr, (long long)CS * CS, CS,
         nullptr, 0, 0);
    quant_probs<<<CH * CS, 256, 0, stream>>>(scores, stats,
                                             (long long)b * CH * CS, st + 6,
                                             pq, CS, CS, CS, scale);
    const __bf16* vtb = vt + (size_t)b * CH * CHD * CS;
    gemm(3, pq, (long long)CS * CS, CS, vtb, (long long)CHD * CS, CS, CS, CHD,
         CS, CH, nullptr, nullptr, nullptr, 0, CD, attn, b, CS);
  }
  gemm(1, attn, 0, CD, Wo1t, 0, CD, CBS, CD, CD, 1, out, out, bo1, 0, CD,
       nullptr, 0, 0);

  // ========================= cross-attention =========================
  ln_bf16<<<CBS, 256, 0, stream>>>(out, ln2g, ln2b, lnbuf, CD);
  gemm(0, lnbuf, 0, CD, Wq2t, 0, CD, CBS, CD, CD, 1, qf, nullptr, nullptr, 0, CD, nullptr, 0, 0);
  gemm(0, encb, 0, CD, Wk2t, 0, CD, CB * CSE, CD, CD, 1, kf, nullptr, nullptr, 0, CD, nullptr, 0, 0);
  gemm(0, encb, 0, CD, Wv2t, 0, CD, CB * CSE, CD, CD, 1, vf, nullptr, nullptr, 0, CD, nullptr, 0, 0);
  stat_init<<<1, 32, 0, stream>>>(st);
  minmax_k<<<2048, 256, 0, stream>>>(qf, (long long)CBS * CD, st + 0);
  minmax_k<<<2048, 256, 0, stream>>>(kf, (long long)CB * CSE * CD, st + 2);
  minmax_k<<<2048, 256, 0, stream>>>(vf, (long long)CB * CSE * CD, st + 4);
  {
    long long tq = (long long)CB * CH * CS * CHDP;
    quant_pad_qk<<<(unsigned)((tq + 255) / 256), 256, 0, stream>>>(
        qf, st, 0, qp, CB, CS, CH, CHD, CHDP, CD);
    long long tk = (long long)CB * CH * CSE * CHDP;
    quant_pad_qk<<<(unsigned)((tk + 255) / 256), 256, 0, stream>>>(
        kf, st, 2, kp, CB, CSE, CH, CHD, CHDP, CD);
    long long tv = (long long)CB * CH * CHD * CSEP;
    quant_v_t<<<(unsigned)((tv + 255) / 256), 256, 0, stream>>>(
        vf, st, 4, vt, CB, CSE, CSEP, CH, CHD, CD);
  }
  // single shot: scores [B*H, S, 96]
  gemm(0, qp, (long long)CS * CHDP, CHDP, kp, (long long)CSE * CHDP, CHDP, CS,
       CSE, CHDP, CB * CH, scores, nullptr, nullptr, (long long)CS * CSEP, CSEP,
       nullptr, 0, 0);
  sm_stats<<<CB * CH * CS, 256, 0, stream>>>(scores, CSE, CSEP, scale, stats, 0,
                                             st + 6);
  quant_probs<<<CB * CH * CS, 256, 0, stream>>>(scores, stats, 0, st + 6, pq,
                                                CSE, CSEP, CSEP, scale);
  gemm(3, pq, (long long)CS * CSEP, CSEP, vt, (long long)CHD * CSEP, CSEP, CS,
       CHD, CSEP, CB * CH, nullptr, nullptr, nullptr, 0, CD, attn, 0, CS);
  gemm(1, attn, 0, CD, Wo2t, 0, CD, CBS, CD, CD, 1, out, out, bo2, 0, CD,
       nullptr, 0, 0);

  // ========================= GEGLU feed-forward =========================
  ln_bf16<<<CBS, 256, 0, stream>>>(out, ln3g, ln3b, lnbuf, CD);
  for (int mc = 0; mc < CBS / CMC; mc++) {
    const __bf16* Ab = lnbuf + (size_t)mc * CMC * CD;
    gemm(1, Ab, 0, CD, W1t, 0, CD, CMC, 2 * CFF, CD, 1, proj, nullptr, bff1, 0,
         2 * CFF, nullptr, 0, 0);
    geglu_k<<<8192, 256, 0, stream>>>(proj, gbuf, CMC, CFF);
    float* ob = out + (size_t)mc * CMC * CD;
    gemm(1, gbuf, 0, CFF, W2t, 0, CFF, CMC, CD, CFF, 1, ob, ob, bff2, 0, CD,
         nullptr, 0, 0);
  }
}
